// MultiHeadAttention_5488968204518
// MI455X (gfx1250) — matrix operand order repaired and checked by reading, not compiled, not run on hardware
//
#include <hip/hip_runtime.h>
#include <hip/hip_bf16.h>
#include <math.h>

typedef __bf16 bf16;
typedef __attribute__((ext_vector_type(16))) __bf16 v16bf;
typedef __attribute__((ext_vector_type(8)))  __bf16 v8bf;
typedef __attribute__((ext_vector_type(8)))  float  v8f;

__device__ __forceinline__ v8f zero_v8f() {
  v8f z;
#pragma unroll
  for (int i = 0; i < 8; ++i) z[i] = 0.0f;
  return z;
}
__device__ __forceinline__ v16bf zero_v16bf() {
  v16bf z;
#pragma unroll
  for (int i = 0; i < 16; ++i) z[i] = (bf16)0.0f;
  return z;
}

// CDNA5 async copy: global -> LDS, 16B per lane, tracked by ASYNCcnt.
// lds_off = wave-relative LDS byte address (low 32 bits of generic pointer).
__device__ __forceinline__ void async_ld_b128(unsigned lds_off, const bf16* gptr) {
  asm volatile("global_load_async_to_lds_b128 %0, %1, off"
               :: "v"(lds_off), "v"((unsigned long long)(size_t)gptr)
               : "memory");
}
__device__ __forceinline__ void wait_async0() {
  asm volatile("s_wait_asynccnt 0x0" ::: "memory");
}

// ---------------------------------------------------------------------------
// fp32 -> bf16 conversion
// ---------------------------------------------------------------------------
__global__ __launch_bounds__(256) void cvt_bf16_kernel(const float* __restrict__ src,
                                                       bf16* __restrict__ dst, int n) {
  int i = blockIdx.x * 256 + threadIdx.x;
  if (i < n) dst[i] = (bf16)src[i];
}

// ---------------------------------------------------------------------------
// C[M,N] = A[M,K] * W[N,K]^T + bias[N]   (torch nn.Linear semantics)
// 128x128 block tile, 8 waves (4 along M x 2 along N), K-step 32, bf16 WMMA.
// Double-buffered LDS filled by async global->LDS copies (ASYNCcnt pipeline).
// ---------------------------------------------------------------------------
template <bool F32OUT>
__global__ __launch_bounds__(256) void gemm_bias_kernel(const bf16* __restrict__ A,
                                                        const bf16* __restrict__ W,
                                                        const float* __restrict__ bias,
                                                        void* __restrict__ outp,
                                                        int M, int N, int K) {
  __shared__ __align__(32) bf16 As[2][128][32];
  __shared__ __align__(32) bf16 Bs[2][128][32];
  const int tid  = threadIdx.x;
  const int lane = tid & 31;
  const int w    = tid >> 5;
  const int wm   = w & 3;          // wave's 32-row strip
  const int wn   = w >> 2;         // wave's 64-col strip
  const int r    = lane & 15;
  const int hh   = lane >> 4;
  const int m0   = blockIdx.x * 128;
  const int n0   = blockIdx.y * 128;
  const int ldr  = tid >> 1;       // 0..127 tile row loaded by this thread
  const int ldc  = (tid & 1) * 16; // 0 or 16 (16 bf16 = 32B per thread)

  v8f acc[2][4];
#pragma unroll
  for (int a = 0; a < 2; ++a)
#pragma unroll
    for (int b = 0; b < 4; ++b) acc[a][b] = zero_v8f();

  const int nt = K >> 5;  // K-tiles of 32

  // prologue: async-fill buffer 0 with tile 0 (4 async instrs per wave)
  {
    const bf16* ga = &A[(size_t)(m0 + ldr) * K + ldc];
    const bf16* gw = &W[(size_t)(n0 + ldr) * K + ldc];
    async_ld_b128((unsigned)(size_t)&As[0][ldr][ldc],     ga);
    async_ld_b128((unsigned)(size_t)&As[0][ldr][ldc + 8], ga + 8);
    async_ld_b128((unsigned)(size_t)&Bs[0][ldr][ldc],     gw);
    async_ld_b128((unsigned)(size_t)&Bs[0][ldr][ldc + 8], gw + 8);
  }

  for (int t = 0; t < nt; ++t) {
    const int cur = t & 1;
    wait_async0();      // my fills of buf[cur] are complete
    __syncthreads();    // everyone's fills complete; prior readers of buf[cur^1] done

    if (t + 1 < nt) {   // kick off next tile into the other buffer (overlaps WMMAs)
      const int k1 = (t + 1) << 5;
      const bf16* ga = &A[(size_t)(m0 + ldr) * K + k1 + ldc];
      const bf16* gw = &W[(size_t)(n0 + ldr) * K + k1 + ldc];
      async_ld_b128((unsigned)(size_t)&As[cur ^ 1][ldr][ldc],     ga);
      async_ld_b128((unsigned)(size_t)&As[cur ^ 1][ldr][ldc + 8], ga + 8);
      async_ld_b128((unsigned)(size_t)&Bs[cur ^ 1][ldr][ldc],     gw);
      async_ld_b128((unsigned)(size_t)&Bs[cur ^ 1][ldr][ldc + 8], gw + 8);
    }

    // A fragments: lane (r,hh) holds row r, K = hh*16..hh*16+15 (contiguous)
    union { v16bf v; v8bf h2[2]; } af[2];
    v16bf bfrag[4];
#pragma unroll
    for (int fm = 0; fm < 2; ++fm) {
      const int row = wm * 32 + fm * 16 + r;
      af[fm].h2[0] = *(const v8bf*)&As[cur][row][hh * 16];
      af[fm].h2[1] = *(const v8bf*)&As[cur][row][hh * 16 + 8];
    }
    // B fragments: lane (r,hh) holds column r, K = hh*16..hh*16+15 (contiguous)
#pragma unroll
    for (int fn = 0; fn < 4; ++fn)
      bfrag[fn] = *(const v16bf*)&Bs[cur][wn * 64 + fn * 16 + r][hh * 16];

#pragma unroll
    for (int fm = 0; fm < 2; ++fm)
#pragma unroll
      for (int fn = 0; fn < 4; ++fn)
        acc[fm][fn] = __builtin_amdgcn_wmma_f32_16x16x32_bf16(
            false, af[fm].v, false, bfrag[fn], (short)0, acc[fm][fn], false, false);
  }

  // epilogue: bias add (f32) + store. C layout: reg i -> row i+8*hh, col r.
#pragma unroll
  for (int fn = 0; fn < 4; ++fn) {
    const int col = n0 + wn * 64 + fn * 16 + r;
    const float bv = bias[col];
#pragma unroll
    for (int fm = 0; fm < 2; ++fm) {
#pragma unroll
      for (int i = 0; i < 8; ++i) {
        const int row = m0 + wm * 32 + fm * 16 + i + hh * 8;
        const float v = acc[fm][fn][i] + bv;
        if (F32OUT) ((float*)outp)[(size_t)row * N + col] = v;
        else        ((bf16*)outp)[(size_t)row * N + col]  = (bf16)v;
      }
    }
  }
}

// ---------------------------------------------------------------------------
// Flash attention: head_dim = 16, 64 chunks x 2 batch, 32 keys per iteration.
// One wave per (chunk, batch, 16-query tile); P/V^T slabs are wave-private LDS
// (same-wave DS ordering -> no barriers). PV WMMA uses the full K=32.
// Output scattered directly into the torch reshape-scrambled layout.
// ---------------------------------------------------------------------------
__global__ __launch_bounds__(256) void attn_kernel(const bf16* __restrict__ Q,
                                                   const bf16* __restrict__ Kc,
                                                   const bf16* __restrict__ V,
                                                   bf16* __restrict__ O) {
  constexpr int S = 1024, F = 1024;
  __shared__ __align__(32) bf16 Psc[8][16][32]; // per-wave P (16 rows x 32 keys)
  __shared__ __align__(32) bf16 Vt[8][16][32];  // per-wave V^T (16 dims x 32 keys)
  const int tid  = threadIdx.x;
  const int lane = tid & 31;
  const int w    = tid >> 5;
  const int r    = lane & 15;
  const int hh   = lane >> 4;
  const int task = blockIdx.x * 8 + w;
  const int p    = task >> 6;  // (chunk, batch) pair
  const int qt   = task & 63;  // 16-row query tile
  const int c    = p >> 1;
  const int bb   = p & 1;

  const bf16* qb = Q  + ((size_t)(bb * S + qt * 16)) * F + c * 16;
  const bf16* kb = Kc + ((size_t)(bb * S)) * F + c * 16;
  const bf16* vb = V  + ((size_t)(bb * S)) * F + c * 16;

  // Q fragment (A-matrix 16x32; head_dim=16 -> upper K half zero)
  union { v16bf v; v8bf h2[2]; } aq;
  aq.v = hh ? zero_v16bf() : *(const v16bf*)(qb + (size_t)r * F);

  float mrow[8], lrow[8];
#pragma unroll
  for (int i = 0; i < 8; ++i) { mrow[i] = -1e30f; lrow[i] = 0.0f; }
  v8f oacc = zero_v8f();
  const float scale = 0.25f; // 1/sqrt(16)

  for (int j0 = 0; j0 < S; j0 += 32) {
    // K fragments (B): lane column = key, 16 contiguous dims; dim pad -> 0
    const v16bf bk0 = hh ? zero_v16bf() : *(const v16bf*)(kb + (size_t)(j0 + r) * F);
    const v16bf bk1 = hh ? zero_v16bf() : *(const v16bf*)(kb + (size_t)(j0 + 16 + r) * F);
    __builtin_prefetch(kb + (size_t)(j0 + 32 + lane) * F, 0, 1);
    __builtin_prefetch(vb + (size_t)(j0 + 32 + lane) * F, 0, 1);
    v8f s0 = __builtin_amdgcn_wmma_f32_16x16x32_bf16(false, aq.v, false, bk0, (short)0,
                                                     zero_v8f(), false, false);
    v8f s1 = __builtin_amdgcn_wmma_f32_16x16x32_bf16(false, aq.v, false, bk1, (short)0,
                                                     zero_v8f(), false, false);
    // online softmax over 32 keys: reg i <-> row i+8*hh within each 16-lane half
    float corr[8]; float p0[8], p1[8];
#pragma unroll
    for (int i = 0; i < 8; ++i) {
      const float a0 = s0[i] * scale, a1 = s1[i] * scale;
      float mx = fmaxf(a0, a1);
#pragma unroll
      for (int d = 1; d < 16; d <<= 1) mx = fmaxf(mx, __shfl_xor(mx, d, 32));
      const float mnew = fmaxf(mrow[i], mx);
      p0[i] = __expf(a0 - mnew);
      p1[i] = __expf(a1 - mnew);
      float sum = p0[i] + p1[i];
#pragma unroll
      for (int d = 1; d < 16; d <<= 1) sum += __shfl_xor(sum, d, 32);
      corr[i] = __expf(mrow[i] - mnew);
      lrow[i] = lrow[i] * corr[i] + sum;
      mrow[i] = mnew;
    }
    // stage P (D-layout -> A-layout) and V^T in this wave's private LDS slabs
#pragma unroll
    for (int i = 0; i < 8; ++i) {
      Psc[w][i + 8 * hh][r]      = (bf16)p0[i];
      Psc[w][i + 8 * hh][16 + r] = (bf16)p1[i];
    }
    {
      const v16bf vrow = *(const v16bf*)(vb + (size_t)(j0 + lane) * F);
#pragma unroll
      for (int d = 0; d < 16; ++d) Vt[w][d][lane] = vrow[d];
    }
    // same-wave DS ops are in-order; compiler inserts s_wait_dscnt for the loads
    union { v16bf v; v8bf h2[2]; } ap;
    ap.h2[0] = *(const v8bf*)&Psc[w][r][hh * 16];
    ap.h2[1] = *(const v8bf*)&Psc[w][r][hh * 16 + 8];
    const v16bf bvv = *(const v16bf*)&Vt[w][r][hh * 16];
    v8f cin;
#pragma unroll
    for (int i = 0; i < 8; ++i) cin[i] = oacc[i] * corr[i];
    oacc = __builtin_amdgcn_wmma_f32_16x16x32_bf16(false, ap.v, false, bvv, (short)0,
                                                   cin, false, false);
  }

  // normalize + scatter into the torch-scrambled layout
#pragma unroll
  for (int i = 0; i < 8; ++i) {
    const float val = oacc[i] / lrow[i];
    const int s2 = qt * 16 + i + 8 * hh;
    const unsigned flat = (((unsigned)(c * 2 + bb) * 1024u + (unsigned)s2) * 16u + (unsigned)r);
    const unsigned hp  = flat >> 17;           // H index after reinterpret
    const unsigned rem = flat & 131071u;
    const unsigned bo  = rem >> 16;            // batch
    const unsigned so  = (rem & 65535u) >> 6;  // seq
    const unsigned cp  = rem & 63u;            // hc index
    O[((size_t)bo * 1024 + so) * 1024 + hp * 64 + cp] = (bf16)val;
  }
}

// ---------------------------------------------------------------------------
extern "C" void kernel_launch(void* const* d_in, const int* in_sizes, int n_in,
                              void* d_out, int out_size, void* d_ws, size_t ws_size,
                              hipStream_t stream) {
  (void)in_sizes; (void)n_in; (void)out_size; (void)ws_size;
  const int Fdim = 1024, Sdim = 1024, Bdim = 2;
  const int M = Bdim * Sdim;                  // 2048
  const size_t NX = (size_t)M * Fdim;         // activation elems
  const size_t NW = (size_t)Fdim * Fdim;      // weight elems

  const float* x = (const float*)d_in[0];
  // dict order: x, then (w,b) for [wq, wk, wv, vq, vk, vv, wo]
  const float* w_ptr[7] = { (const float*)d_in[1],  (const float*)d_in[3],
                            (const float*)d_in[5],  (const float*)d_in[7],
                            (const float*)d_in[9],  (const float*)d_in[11],
                            (const float*)d_in[13] };
  const float* b_ptr[7] = { (const float*)d_in[2],  (const float*)d_in[4],
                            (const float*)d_in[6],  (const float*)d_in[8],
                            (const float*)d_in[10], (const float*)d_in[12],
                            (const float*)d_in[14] };

  // workspace layout (bf16): X | 7 weights | T1 | Q | K | V | O  (~38 MB)
  bf16* Xbf = (bf16*)d_ws;
  bf16* Wbf[7];
  for (int i = 0; i < 7; ++i) Wbf[i] = Xbf + NX + (size_t)i * NW;
  bf16* T1 = Xbf + NX + 7 * NW;
  bf16* Qb = T1 + NX;
  bf16* Kb = Qb + NX;
  bf16* Vb = Kb + NX;
  bf16* Ob = Vb + NX;

  cvt_bf16_kernel<<<(int)((NX + 255) / 256), 256, 0, stream>>>(x, Xbf, (int)NX);
  for (int i = 0; i < 7; ++i)
    cvt_bf16_kernel<<<(int)((NW + 255) / 256), 256, 0, stream>>>(w_ptr[i], Wbf[i], (int)NW);

  dim3 g(M / 128, Fdim / 128);  // (16, 8)
  // q = (x Wq^T + bq) Vq^T + bvq ; same for k, v
  gemm_bias_kernel<false><<<g, 256, 0, stream>>>(Xbf, Wbf[0], b_ptr[0], T1, M, Fdim, Fdim);
  gemm_bias_kernel<false><<<g, 256, 0, stream>>>(T1,  Wbf[3], b_ptr[3], Qb, M, Fdim, Fdim);
  gemm_bias_kernel<false><<<g, 256, 0, stream>>>(Xbf, Wbf[1], b_ptr[1], T1, M, Fdim, Fdim);
  gemm_bias_kernel<false><<<g, 256, 0, stream>>>(T1,  Wbf[4], b_ptr[4], Kb, M, Fdim, Fdim);
  gemm_bias_kernel<false><<<g, 256, 0, stream>>>(Xbf, Wbf[2], b_ptr[2], T1, M, Fdim, Fdim);
  gemm_bias_kernel<false><<<g, 256, 0, stream>>>(T1,  Wbf[5], b_ptr[5], Vb, M, Fdim, Fdim);

  attn_kernel<<<1024, 256, 0, stream>>>(Qb, Kb, Vb, Ob);

  gemm_bias_kernel<true><<<g, 256, 0, stream>>>(Ob, Wbf[6], b_ptr[6], d_out, M, Fdim, Fdim);
}
